// MultiScaleMultiHeadAttention_61984968015923
// MI455X (gfx1250) — compile-verified, matrix operand order checked by reading, not run on hardware
//
#include <hip/hip_runtime.h>
#include <hip/hip_bf16.h>

// MI455X / gfx1250, wave32. WMMA bf16 16x16x32 with f32 accumulate.
// B tiles staged through LDS (bf16) with cooperative coalesced loads.

typedef __attribute__((ext_vector_type(16))) __bf16 v16bf;
typedef __attribute__((ext_vector_type(8)))  __bf16 v8bf;
typedef __attribute__((ext_vector_type(8)))  float  v8f;

constexpr int kB  = 8;
constexpr int kN  = 512;
constexpr int kH  = 256;
constexpr int kNH = 8;
constexpr int kDK = 32;
constexpr float kNEG = -1e12f;

__device__ __forceinline__ int imin(int a, int b) { return a < b ? a : b; }

// ---- A fragment (global, ISA 7.12.2 layout, wave32) ------------------------
// A 16x32 bf16: lane = {m = lane&15, khalf = lane>>4}; elements 0..7 hold
// K = kb..kb+7, elements 8..15 hold K = kb+16..kb+23 (kb = k0 + 8*khalf).
__device__ __forceinline__ v16bf load_a_frag(const float* __restrict__ ar) {
  const float4 p0 = *(const float4*)(ar);
  const float4 p1 = *(const float4*)(ar + 4);
  const float4 p2 = *(const float4*)(ar + 16);
  const float4 p3 = *(const float4*)(ar + 20);
  v16bf a;
  a[0]=(__bf16)p0.x; a[1]=(__bf16)p0.y; a[2]=(__bf16)p0.z; a[3]=(__bf16)p0.w;
  a[4]=(__bf16)p1.x; a[5]=(__bf16)p1.y; a[6]=(__bf16)p1.z; a[7]=(__bf16)p1.w;
  a[8]=(__bf16)p2.x; a[9]=(__bf16)p2.y; a[10]=(__bf16)p2.z; a[11]=(__bf16)p2.w;
  a[12]=(__bf16)p3.x; a[13]=(__bf16)p3.y; a[14]=(__bf16)p3.z; a[15]=(__bf16)p3.w;
  return a;
}

// ---- B fragment from LDS tile (n-major bt[64][32]) -------------------------
// lane column fixed; two aligned 16B runs -> ds_load_b128 x2.
__device__ __forceinline__ v16bf load_b_lds(const __bf16* __restrict__ btrow,
                                            int lhalf) {
  const v8bf lo = *(const v8bf*)(btrow + 8 * lhalf);
  const v8bf hi = *(const v8bf*)(btrow + 16 + 8 * lhalf);
  return __builtin_shufflevector(lo, hi, 0,1,2,3,4,5,6,7,8,9,10,11,12,13,14,15);
}

// ---- generic batched WMMA GEMM:  C = alpha * A@B + bias  (+relu / +mask) ---
// Batch offset = (batch>>bshift)*s_hi + (batch&((1<<bshift)-1))*s_lo per tensor.
// A: [M,K] row-major (col stride 1). B: element [k][n] at k*ldbk + n*ldbn.
// maskp (optional): int [Bhi, M, Nc]; 0 -> write -1e12 (applied after alpha).
__global__ __launch_bounds__(256) void gemm_wmma_bf16(
    const float* __restrict__ A,  long sa_hi, long sa_lo, int lda,
    const float* __restrict__ Bm, long sb_hi, long sb_lo, int ldbk, int ldbn,
    float* __restrict__ C,        long sc_hi, long sc_lo, int ldc,
    const float* __restrict__ bias,
    const int* __restrict__ maskp,
    int bshift, int M, int Nc, int K, float alpha, int relu)
{
  __shared__ __bf16 bt[64][32];            // B tile, n-major, 4 KB

  const int t     = threadIdx.x;
  const int lane  = t & 31;
  const int wid   = t >> 5;                // 8 waves
  const int wm    = wid >> 1;              // 0..3
  const int wn    = wid & 1;               // 0..1
  const int nblk  = blockIdx.x * 64;
  const int m0    = blockIdx.y * 128 + wm * 32;
  const int n0    = nblk + wn * 32;
  const int batch = blockIdx.z;
  const int bhi   = batch >> bshift;
  const int blo   = batch & ((1 << bshift) - 1);

  const float* Ab = A  + (long)bhi * sa_hi + (long)blo * sa_lo;
  const float* Bb = Bm + (long)bhi * sb_hi + (long)blo * sb_lo;
  float*       Cb = C  + (long)bhi * sc_hi + (long)blo * sc_lo;

  const int lrow  = lane & 15;
  const int lhalf = lane >> 4;

  // branch-free clamps keep addresses in-bounds without perturbing EXEC
  const int mr0 = imin(m0 + lrow,      M - 1);
  const int mr1 = imin(m0 + 16 + lrow, M - 1);

  // staging coordinates: thread -> (n_local 0..63, k-octet 0..3)
  const int snl = t >> 2;
  const int sk8 = (t & 3) * 8;
  const int sng = imin(nblk + snl, Nc - 1);

  v8f acc00 = {}, acc01 = {}, acc10 = {}, acc11 = {};

  for (int k0 = 0; k0 < K; k0 += 32) {
    __syncthreads();
    {   // cooperative B stage: 32k x 64n f32 -> bf16, one ds_store_b128/thread
      const float* src = Bb + (long)sng * ldbn + (long)(k0 + sk8) * ldbk;
      v8bf v;
      if (ldbk == 1) {                       // contiguous K (scores: B = k^T)
        const float4 f0 = *(const float4*)(src);
        const float4 f1 = *(const float4*)(src + 4);
        v[0]=(__bf16)f0.x; v[1]=(__bf16)f0.y; v[2]=(__bf16)f0.z; v[3]=(__bf16)f0.w;
        v[4]=(__bf16)f1.x; v[5]=(__bf16)f1.y; v[6]=(__bf16)f1.z; v[7]=(__bf16)f1.w;
      } else {                               // strided K (weights, V)
#pragma unroll
        for (int i = 0; i < 8; ++i) v[i] = (__bf16)src[(long)i * ldbk];
      }
      *(v8bf*)(&bt[snl][sk8]) = v;
    }
    __syncthreads();

    const int kb = k0 + 8 * lhalf;
    const float* ar0 = Ab + (long)mr0 * lda + kb;
    const float* ar1 = Ab + (long)mr1 * lda + kb;
    if (k0 + 32 < K) {                       // gfx1250 global_prefetch_b8
      __builtin_prefetch(ar0 + 32, 0, 0);
      __builtin_prefetch(ar1 + 32, 0, 0);
    }
    const v16bf a0 = load_a_frag(ar0);
    const v16bf a1 = load_a_frag(ar1);
    const v16bf b0 = load_b_lds(&bt[wn * 32 + lrow][0], lhalf);
    const v16bf b1 = load_b_lds(&bt[wn * 32 + 16 + lrow][0], lhalf);

    acc00 = __builtin_amdgcn_wmma_f32_16x16x32_bf16(false, a0, false, b0, (short)0, acc00, false, false);
    acc01 = __builtin_amdgcn_wmma_f32_16x16x32_bf16(false, a0, false, b1, (short)0, acc01, false, false);
    acc10 = __builtin_amdgcn_wmma_f32_16x16x32_bf16(false, a1, false, b0, (short)0, acc10, false, false);
    acc11 = __builtin_amdgcn_wmma_f32_16x16x32_bf16(false, a1, false, b1, (short)0, acc11, false, false);
  }

  // epilogue: C layout — lane 0-15: row = r; lane 16-31: row = 8 + r
  v8f accs[2][2] = {{acc00, acc01}, {acc10, acc11}};
#pragma unroll
  for (int i = 0; i < 2; ++i) {
#pragma unroll
    for (int j = 0; j < 2; ++j) {
      const int ncol = n0 + 16 * j + lrow;
      if (ncol >= Nc) continue;
      const float bv = bias ? bias[ncol] : 0.0f;
#pragma unroll
      for (int r = 0; r < 8; ++r) {
        const int mrow = m0 + 16 * i + 8 * lhalf + r;
        if (mrow >= M) continue;
        float v = accs[i][j][r] * alpha + bv;
        if (relu) v = fmaxf(v, 0.0f);
        if (maskp && maskp[((long)bhi * M + mrow) * Nc + ncol] == 0) v = kNEG;
        Cb[(long)mrow * ldc + ncol] = v;
      }
    }
  }
}

// ---- per-bar distance-masked softmax over rows of [B,nh,N,N] ---------------
__global__ __launch_bounds__(256) void softmax_bar(
    const float* __restrict__ sc, const float* __restrict__ dist,
    float* __restrict__ out, float* __restrict__ out2, float bar)
{
  __shared__ float red[256];
  const int t = threadIdx.x;
  const int m = blockIdx.x, h = blockIdx.y, b = blockIdx.z;
  const long srow = ((((long)b * kNH) + h) * kN + m) * kN;
  const long drow = ((long)b * kN + m) * kN;
  const int n0 = t, n1 = t + 256;

  float s0 = sc[srow + n0];
  float s1 = sc[srow + n1];
  const bool d0 = (m == 0) || (n0 == 0) || (dist[drow + n0] < bar);
  const bool d1 = (m == 0) || (dist[drow + n1] < bar);   // n1 >= 256, never col 0
  s0 = d0 ? s0 : kNEG;
  s1 = d1 ? s1 : kNEG;

  red[t] = fmaxf(s0, s1);
  __syncthreads();
  for (int off = 128; off; off >>= 1) { if (t < off) red[t] = fmaxf(red[t], red[t + off]); __syncthreads(); }
  const float mx = red[0];
  __syncthreads();

  const float e0 = __expf(s0 - mx);
  const float e1 = __expf(s1 - mx);
  red[t] = e0 + e1;
  __syncthreads();
  for (int off = 128; off; off >>= 1) { if (t < off) red[t] += red[t + off]; __syncthreads(); }
  const float inv = 1.0f / red[0];

  out[srow + n0] = e0 * inv;
  out[srow + n1] = e1 * inv;
  if (out2) { out2[srow + n0] = e0 * inv; out2[srow + n1] = e1 * inv; }
}

// ---- residual + LayerNorm over H=256 ---------------------------------------
__global__ __launch_bounds__(256) void add_ln(
    const float* __restrict__ hb, const float* __restrict__ x,
    const float* __restrict__ gamma, const float* __restrict__ beta,
    float* __restrict__ y)
{
  __shared__ float red[256];
  const int t = threadIdx.x;
  const long m = blockIdx.x;
  const float v = hb[m * kH + t] + x[m * kH + t];

  red[t] = v;
  __syncthreads();
  for (int off = 128; off; off >>= 1) { if (t < off) red[t] += red[t + off]; __syncthreads(); }
  const float mu = red[0] * (1.0f / kH);
  __syncthreads();

  const float d = v - mu;
  red[t] = d * d;
  __syncthreads();
  for (int off = 128; off; off >>= 1) { if (t < off) red[t] += red[t + off]; __syncthreads(); }
  const float var = red[0] * (1.0f / kH);

  y[m * kH + t] = d * rsqrtf(var + 1e-6f) * gamma[t] + beta[t];
}

extern "C" void kernel_launch(void* const* d_in, const int* in_sizes, int n_in,
                              void* d_out, int out_size, void* d_ws, size_t ws_size,
                              hipStream_t stream) {
  (void)in_sizes; (void)n_in; (void)out_size; (void)ws_size;
  const float* x    = (const float*)d_in[0];
  const float* dist = (const float*)d_in[1];
  const int*   mask = (const int*)  d_in[2];
  const float* Wq   = (const float*)d_in[3];  const float* bq  = (const float*)d_in[4];
  const float* Wk   = (const float*)d_in[5];  const float* bk  = (const float*)d_in[6];
  const float* Wv   = (const float*)d_in[7];  const float* bv  = (const float*)d_in[8];
  const float* Wa   = (const float*)d_in[9];  const float* ba  = (const float*)d_in[10];
  const float* Ws1  = (const float*)d_in[11]; const float* bs1 = (const float*)d_in[12];
  const float* Ws2  = (const float*)d_in[13]; const float* bs2 = (const float*)d_in[14];
  const float* gamma = (const float*)d_in[15];
  const float* beta  = (const float*)d_in[16];

  // workspace (f32 elements); high-water ~92 MB with region reuse
  float* ws = (float*)d_ws;
  float* qf     = ws;                 // [B,N,H]   1,048,576
  float* kf     = ws + 1048576;       // [B,N,H]
  float* vf     = ws + 2097152;       // [B,N,H]
  float* scores = ws + 3145728;       // [B,nh,N,N] 16,777,216 (mask applied)
  float* cat    = ws + 19922944;      // [B*N, 3H]  3,145,728
  float* o      = qf;                 // reuse: qf dead after scores GEMM
  float* ff1    = vf;                 // reuse: vf dead after last attn@v
  float* hb     = kf;                 // reuse: kf dead after scores GEMM

  float* out       = (float*)d_out;
  float* attn_last = out + 1048576;                      // [B,nh,N,N]
  float* stack     = out + 1048576 + 16777216;           // [3,B,nh,N,N]

  const dim3 blk(256);
  const int  MN = kB * kN;                               // 4096
  const long NH2 = (long)kN * kH;                        // per-b stride in q/k/v
  const long NN  = (long)kN * kN;

  // Q/K/V projections: [4096,256] @ [256,256] + bias
  const dim3 gproj(kH / 64, MN / 128, 1);
  gemm_wmma_bf16<<<gproj, blk, 0, stream>>>(x, 0, 0, kH,  Wq, 0, 0, kH, 1,
      qf, 0, 0, kH,  bq, nullptr, 0, MN, kH, kH, 1.0f, 0);
  gemm_wmma_bf16<<<gproj, blk, 0, stream>>>(x, 0, 0, kH,  Wk, 0, 0, kH, 1,
      kf, 0, 0, kH,  bk, nullptr, 0, MN, kH, kH, 1.0f, 0);
  gemm_wmma_bf16<<<gproj, blk, 0, stream>>>(x, 0, 0, kH,  Wv, 0, 0, kH, 1,
      vf, 0, 0, kH,  bv, nullptr, 0, MN, kH, kH, 1.0f, 0);

  // scores = (q @ k^T)/sqrt(dk), padding mask fused (mask==0 -> -1e12)
  // batch = b*8+h: A/B base += b*(N*H) + h*dk ; C base += b*8*N*N + h*N*N
  const dim3 gsc(kN / 64, kN / 128, kB * kNH);
  gemm_wmma_bf16<<<gsc, blk, 0, stream>>>(
      qf, NH2, kDK, kH,
      kf, NH2, kDK, /*ldbk=*/1, /*ldbn=*/kH,      // B[k][n] = k_h[n][k]
      scores, (long)kNH * NN, NN, kN,
      nullptr, mask, /*bshift=*/3, kN, kN, kDK, 0.17677669529663687f, 0);

  const float bars[3] = {0.3f, 0.6f, 1.0f};
  for (int ib = 0; ib < 3; ++ib) {
    float* attn = stack + (long)ib * kB * kNH * NN;
    const dim3 gsm(kN, kNH, kB);
    softmax_bar<<<gsm, blk, 0, stream>>>(scores, dist, attn,
                                         (ib == 2) ? attn_last : nullptr, bars[ib]);

    // o = attn @ v : per (b,h) [512,512] @ [512,32]; attn read f32 from d_out
    const dim3 gav(1, kN / 128, kB * kNH);
    gemm_wmma_bf16<<<gav, blk, 0, stream>>>(
        attn, (long)kNH * NN, NN, kN,
        vf,   NH2, kDK, /*ldbk=*/kH, /*ldbn=*/1,
        o,    NH2, kDK, kH,
        nullptr, nullptr, 3, kN, kDK, kN, 1.0f, 0);

    // outs[ib] = o @ Wa + ba  -> cat[:, ib*H : (ib+1)*H]
    gemm_wmma_bf16<<<gproj, blk, 0, stream>>>(o, 0, 0, kH,  Wa, 0, 0, kH, 1,
        cat + ib * kH, 0, 0, 3 * kH,  ba, nullptr, 0, MN, kH, kH, 1.0f, 0);
  }

  // FF1: relu(cat @ Ws1 + bs1), K=768
  gemm_wmma_bf16<<<gproj, blk, 0, stream>>>(cat, 0, 0, 3 * kH,  Ws1, 0, 0, kH, 1,
      ff1, 0, 0, kH,  bs1, nullptr, 0, MN, kH, 3 * kH, 1.0f, 1);
  // FF2: h = ff1 @ Ws2 + bs2
  gemm_wmma_bf16<<<gproj, blk, 0, stream>>>(ff1, 0, 0, kH,  Ws2, 0, 0, kH, 1,
      hb, 0, 0, kH,  bs2, nullptr, 0, MN, kH, kH, 1.0f, 0);

  // y = LN(h + x) * gamma + beta
  add_ln<<<dim3(MN), blk, 0, stream>>>(hb, x, gamma, beta, out);
}
